// SemGraphConv_45784351375439
// MI455X (gfx1250) — matrix operand order repaired and checked by reading, not compiled, not run on hardware
//
#include <hip/hip_runtime.h>
#include <hip/hip_bf16.h>
#include <math.h>

typedef __attribute__((ext_vector_type(16))) __bf16 v16bf;
typedef __attribute__((ext_vector_type(8)))  float  v8f;

#define BATCH  131072
#define KDIM   512      // 16 * 32  (n'' , f)
#define NDIM   512      // 16 * 32  (m , o)
#define WP_BYTES ((size_t)KDIM * NDIM * 2)   // packed bf16 weight matrix

// ---------------------------------------------------------------------------
// Kernel 1: diag[i] = softmax(where(adj>0, e, -9e15), axis=1)[i,i]
// ---------------------------------------------------------------------------
__global__ void sgc_diag_kernel(const float* __restrict__ adj,
                                const float* __restrict__ e,
                                float* __restrict__ diag) {
    int i = threadIdx.x;
    if (i < 16) {
        float mx = -3.4e38f;
        float lii = 0.f;
        #pragma unroll
        for (int j = 0; j < 16; ++j) {
            float v = (adj[i * 16 + j] > 0.f) ? e[i * 16 + j] : -9.0e15f;
            if (j == i) lii = v;
            mx = fmaxf(mx, v);
        }
        float s = 0.f;
        #pragma unroll
        for (int j = 0; j < 16; ++j) {
            float v = (adj[i * 16 + j] > 0.f) ? e[i * 16 + j] : -9.0e15f;
            s += expf(v - mx);
        }
        diag[i] = expf(lii - mx) / s;
    }
}

// ---------------------------------------------------------------------------
// Kernel 2: fold basis/W_e/W_2/W_3/diag into one 512x512 matrix M and store it
// bf16, already permuted into a B-fragment layout with contiguous k, matching A:
//   B-frag (16x16x32 bf16): lane l holds column N = (l&15);
//     element e of v16bf  <->  K = e + ((l>=16) ? 16 : 0)
// Packed index: (((ct*16 + ks)*32 + l)*16 + e), ct = N/16 tile, ks = K/32 step.
// ---------------------------------------------------------------------------
__global__ void sgc_fold_weights_kernel(const float* __restrict__ basis,
                                        const float* __restrict__ W_e,
                                        const float* __restrict__ W_2,
                                        const float* __restrict__ W_3,
                                        const float* __restrict__ diag,
                                        __bf16* __restrict__ wp) {
    int tid = blockIdx.x * blockDim.x + threadIdx.x;   // 0 .. 262143
    int el = tid & 15;
    int l  = (tid >> 4) & 31;
    int ks = (tid >> 9) & 15;
    int ct = tid >> 13;                                // 0..31

    int n = ct * 16 + (l & 15);                        // GEMM column j
    int k = ks * 32 + el + ((l >= 16) ? 16 : 0);       // GEMM K index
    int m  = n >> 5, o = n & 31;                       // j = m*32 + o
    int nn = k >> 5, f = k & 31;                       // k = nn*32 + f

    float acc = 0.f;
    // e-block: n,n' in [0,10)  W_e[o*10+n, f*10+n']  (320x320 row-major)
    #pragma unroll
    for (int a = 0; a < 10; ++a) {
        float bm = basis[m * 16 + a];
        const float* wr = W_e + (o * 10 + a) * 320 + f * 10;
        float s = 0.f;
        #pragma unroll
        for (int b = 0; b < 10; ++b) s += wr[b] * basis[nn * 16 + b];
        acc += bm * s;
    }
    // 2-block: n,n' in [10,13)  W_2[o*3+(n-10), f*3+(n'-10)] (96x96)
    #pragma unroll
    for (int a = 0; a < 3; ++a) {
        float bm = basis[m * 16 + 10 + a];
        const float* wr = W_2 + (o * 3 + a) * 96 + f * 3;
        float s = 0.f;
        #pragma unroll
        for (int b = 0; b < 3; ++b) s += wr[b] * basis[nn * 16 + 10 + b];
        acc += bm * s;
    }
    // 3-block: n,n' in [13,16)  W_3[o*3+(n-13), f*3+(n'-13)] (96x96)
    #pragma unroll
    for (int a = 0; a < 3; ++a) {
        float bm = basis[m * 16 + 13 + a];
        const float* wr = W_3 + (o * 3 + a) * 96 + f * 3;
        float s = 0.f;
        #pragma unroll
        for (int b = 0; b < 3; ++b) s += wr[b] * basis[nn * 16 + 13 + b];
        acc += bm * s;
    }
    wp[tid] = (__bf16)(acc * diag[m]);
}

// ---------------------------------------------------------------------------
// Kernel 3: OUT[131072, 512] = X[131072, 512] @ M^T + biasvec
// 256 threads = 8 waves. Block covers 64 rows x 512 cols.
//   wave w: col quarter cq = w&3 (128 cols), row group rg = w>>2 (32 rows),
//   each wave owns TWO 16-row tiles so every B fragment is reused from
//   registers across 2 WMMAs (halves packed-weight refetch traffic).
// A fragment (16x32 bf16) built in-register from f32 x:
//   lane l: row = l&15;  hi = l>>4
//   elems 0..15 <-> K = hi*16 + 0..15    (contiguous 64B -> global_load_b128)
// ---------------------------------------------------------------------------
__global__ void __launch_bounds__(256, 2)
sgc_gemm_kernel(const float* __restrict__ X,
                const __bf16* __restrict__ wp,
                const float* __restrict__ bias,
                float* __restrict__ OUT) {
    const int tid   = threadIdx.x;
    const int wave  = tid >> 5;
    const int lane  = tid & 31;
    const int rg    = wave >> 2;       // 0/1 : which 32-row group
    const int cq    = wave & 3;        // 0..3: which 128-col quarter
    const int r     = lane & 15;
    const int hi    = lane >> 4;

    const size_t rowBase = (size_t)blockIdx.x * 64 + (size_t)rg * 32;
    const int    colBase = cq * 128;

    const float* xrow0 = X + (rowBase + r) * (size_t)KDIM;        // tile 0
    const float* xrow1 = xrow0 + 16 * (size_t)KDIM;               // tile 1

    v8f acc0[8] = {};
    v8f acc1[8] = {};

    #pragma unroll 2
    for (int ks = 0; ks < 16; ++ks) {
        const int kb = ks * 32 + hi * 16;
        __builtin_prefetch(xrow0 + kb + 32, 0, 3);
        __builtin_prefetch(xrow1 + kb + 32, 0, 3);

        const float* p0 = xrow0 + kb;
        float4 a0 = *(const float4*)(p0);
        float4 a1 = *(const float4*)(p0 + 4);
        float4 a2 = *(const float4*)(p0 + 8);
        float4 a3 = *(const float4*)(p0 + 12);
        v16bf A0;
        A0[0]  = (__bf16)a0.x; A0[1]  = (__bf16)a0.y; A0[2]  = (__bf16)a0.z; A0[3]  = (__bf16)a0.w;
        A0[4]  = (__bf16)a1.x; A0[5]  = (__bf16)a1.y; A0[6]  = (__bf16)a1.z; A0[7]  = (__bf16)a1.w;
        A0[8]  = (__bf16)a2.x; A0[9]  = (__bf16)a2.y; A0[10] = (__bf16)a2.z; A0[11] = (__bf16)a2.w;
        A0[12] = (__bf16)a3.x; A0[13] = (__bf16)a3.y; A0[14] = (__bf16)a3.z; A0[15] = (__bf16)a3.w;

        const float* p1 = xrow1 + kb;
        float4 b0 = *(const float4*)(p1);
        float4 b1 = *(const float4*)(p1 + 4);
        float4 b2 = *(const float4*)(p1 + 8);
        float4 b3 = *(const float4*)(p1 + 12);
        v16bf A1;
        A1[0]  = (__bf16)b0.x; A1[1]  = (__bf16)b0.y; A1[2]  = (__bf16)b0.z; A1[3]  = (__bf16)b0.w;
        A1[4]  = (__bf16)b1.x; A1[5]  = (__bf16)b1.y; A1[6]  = (__bf16)b1.z; A1[7]  = (__bf16)b1.w;
        A1[8]  = (__bf16)b2.x; A1[9]  = (__bf16)b2.y; A1[10] = (__bf16)b2.z; A1[11] = (__bf16)b2.w;
        A1[12] = (__bf16)b3.x; A1[13] = (__bf16)b3.y; A1[14] = (__bf16)b3.z; A1[15] = (__bf16)b3.w;

        #pragma unroll
        for (int c = 0; c < 8; ++c) {
            const int ct = cq * 8 + c;
            const v16bf Bf = *(const v16bf*)(wp + ((size_t)((ct * 16 + ks) * 32 + lane) << 4));
            acc0[c] = __builtin_amdgcn_wmma_f32_16x16x32_bf16(
                false, A0, false, Bf, (short)0, acc0[c], false, false);
            acc1[c] = __builtin_amdgcn_wmma_f32_16x16x32_bf16(
                false, A1, false, Bf, (short)0, acc1[c], false, false);
        }
    }

    // Epilogue: C/D layout -> VGPR v, lane: M = v + hi*8, N = lane&15
    const float bb0 = bias[r];
    const float bb1 = bias[r + 16];
    #pragma unroll
    for (int c = 0; c < 8; ++c) {
        const int col  = colBase + c * 16 + r;
        const float ba = (c & 1) ? bb1 : bb0;      // bias[col & 31]
        float* op0 = OUT + (rowBase + (size_t)hi * 8) * NDIM + col;
        float* op1 = op0 + 16 * (size_t)NDIM;
        #pragma unroll
        for (int v = 0; v < 8; ++v) {
            op0[(size_t)v * NDIM] = acc0[c][v] + ba;
            op1[(size_t)v * NDIM] = acc1[c][v] + ba;
        }
    }
}

// ---------------------------------------------------------------------------
extern "C" void kernel_launch(void* const* d_in, const int* in_sizes, int n_in,
                              void* d_out, int out_size, void* d_ws, size_t ws_size,
                              hipStream_t stream) {
    const float* x     = (const float*)d_in[0];   // [131072, 16, 32]
    const float* basis = (const float*)d_in[1];   // [16, 16]
    const float* adj   = (const float*)d_in[2];   // [16, 16]
    const float* e     = (const float*)d_in[3];   // [16, 16]
    const float* W_e   = (const float*)d_in[4];   // [320, 320]
    const float* W_2   = (const float*)d_in[5];   // [96, 96]
    const float* W_3   = (const float*)d_in[6];   // [96, 96]
    const float* bias  = (const float*)d_in[7];   // [32]
    float* out = (float*)d_out;                   // [131072, 16, 32]

    __bf16* wp  = (__bf16*)d_ws;                              // 512 KB packed M
    float* diag = (float*)((char*)d_ws + WP_BYTES);           // 16 floats

    sgc_diag_kernel<<<1, 32, 0, stream>>>(adj, e, diag);

    sgc_fold_weights_kernel<<<(KDIM * NDIM) / 256, 256, 0, stream>>>(
        basis, W_e, W_2, W_3, diag, wp);

    sgc_gemm_kernel<<<BATCH / 64, 256, 0, stream>>>(x, wp, bias, out);
}
